// RVAE_72112500900514
// MI455X (gfx1250) — matrix operand order repaired and checked by reading, not compiled, not run on hardware
//
#include <hip/hip_runtime.h>
#include <hip/hip_bf16.h>

// ---------------------------------------------------------------------------
// RVAE on gfx1250: WMMA-f16 everywhere, f32 accumulation.
//   K1: convert/pad weights to f16, sum LSTM biases
//   K2: Gx = x @ Wih_x^T + (bih_x+bhh_x)        (parallel WMMA GEMM, f16 out)
//   K3: encoder LSTM, backward, persistent 1-WG  (h ping-pong + c state in LDS)
//   K4: z-recurrence (dense tanh, heads, reparam, z-LSTM cell), persistent 1-WG
//   K5: decoder LSTM, forward, persistent 1-WG
//   K6: x_rec = exp(h_dec @ Wd^T + bd)          (parallel WMMA GEMM)
// Register policy: parallel GEMMs run 256-thread blocks with
// __launch_bounds__(256,1) and <=4 accumulator tiles; persistent sequential
// kernels run 512 threads (16 waves, 4/SIMD) with __launch_bounds__(512,1) so
// each wave gets a double VGPR budget -> no scratch spills in the 512-step loop.
// ---------------------------------------------------------------------------

typedef __attribute__((ext_vector_type(16))) _Float16 v16h;
typedef __attribute__((ext_vector_type(8)))  _Float16 v8h;
typedef __attribute__((ext_vector_type(8)))  float    v8f;

#define SEQL  512
#define BATCH 128
#define XD    513
#define XDP   544      // 513 padded to 17*32
#define HD    256
#define ZD    32
#define GD    1024     // 4*HD
#define WDP   528      // 513 padded to 33*16 (rows of Wd)

__device__ __forceinline__ v8f wmma16(v16h a, v16h b, v8f c) {
  return __builtin_amdgcn_wmma_f32_16x16x32_f16(false, a, false, b, (short)0, c,
                                                false, false);
}
__device__ __forceinline__ v8f zero8() {
  v8f z = {0.f, 0.f, 0.f, 0.f, 0.f, 0.f, 0.f, 0.f};
  return z;
}

// A fragment: 16x32 f16 tile of row-major activations (lda in elements).
// ISA layout: lanes 0-15 row m=lane, K {0..7,16..23}; lanes 16-31 K {8..15,24..31}.
__device__ __forceinline__ v16h ldA(const _Float16* base, int lda, int kBase, int lane) {
  int m = lane & 15;
  int koff = (lane & 16) ? 8 : 0;
  const _Float16* p = base + (size_t)m * lda + kBase + koff;
  v8h lo = *(const v8h*)(p);
  v8h hi = *(const v8h*)(p + 16);
  v16h r;
#pragma unroll
  for (int e = 0; e < 8; ++e) { r[e] = lo[e]; r[8 + e] = hi[e]; }
  return r;
}

// B fragment: 32x16 f16 tile of W^T where W is row-major (n_out, K) (ldb = K).
// B[k][n] = W[n][k]; lanes 0-15 hold K {0..7,16..23}, lanes 16-31 hold K {8..15,24..31}.
__device__ __forceinline__ v16h ldB(const _Float16* base, int ldb, int kBase, int lane) {
  int n = lane & 15;
  int koff = (lane & 16) ? 8 : 0;
  const _Float16* p = base + (size_t)n * ldb + kBase + koff;
  v8h lo = *(const v8h*)(p);
  v8h hi = *(const v8h*)(p + 16);
  v16h r;
#pragma unroll
  for (int e = 0; e < 8; ++e) { r[e] = lo[e]; r[8 + e] = hi[e]; }
  return r;
}

__device__ __forceinline__ float sigm(float x) { return 1.f / (1.f + __expf(-x)); }

// ---------------------------------------------------------------------------
// K1a: f32 -> f16 with zero padding (rows and cols)
__global__ void cvt_pad(const float* __restrict__ src, _Float16* __restrict__ dst,
                        int srows, int scols, int drows, int dcols) {
  long total = (long)drows * dcols;
  for (long i = (long)blockIdx.x * blockDim.x + threadIdx.x; i < total;
       i += (long)gridDim.x * blockDim.x) {
    int r = (int)(i / dcols), c = (int)(i % dcols);
    float v = (r < srows && c < scols) ? src[(long)r * scols + c] : 0.f;
    dst[i] = (_Float16)v;
  }
}
// K1b: combined LSTM bias
__global__ void bias_sum(const float* a, const float* b, float* o, int n) {
  int i = blockIdx.x * blockDim.x + threadIdx.x;
  if (i < n) o[i] = a[i] + b[i];
}

// ---------------------------------------------------------------------------
// K2: GxH[(s*B+b), 1024] = f16( x @ Wih_x^T + biasx )   M=65536 N=1024 K=544pad
// grid (512, 16), block 256 (8 waves). Stage 128x544 A panel in LDS as f16.
// K-outer, 4 accumulators/wave: A-frag loaded once per k, reused over 4 N-tiles.
__global__ __launch_bounds__(256, 1) void gemm_gx(
    const float* __restrict__ x, const _Float16* __restrict__ WihxH,
    const float* __restrict__ biasx, _Float16* __restrict__ GxH) {
  extern __shared__ char smem[];
  _Float16* sA = (_Float16*)smem;  // 128 * 544
  int tid = threadIdx.x, lane = tid & 31, w = tid >> 5;
  size_t rowBase = (size_t)blockIdx.x * 128;
  int nBase = blockIdx.y * 64;
  for (int i = tid; i < 128 * XDP; i += 256) {
    int r = i / XDP, c = i % XDP;
    float v = (c < XD) ? x[(rowBase + r) * XD + c] : 0.f;
    sA[i] = (_Float16)v;
  }
  __syncthreads();
  const _Float16* aRow = sA + (size_t)w * 16 * XDP;
  v8f acc[4];
#pragma unroll
  for (int nt = 0; nt < 4; ++nt) acc[nt] = zero8();
  for (int k = 0; k < 17; ++k) {
    v16h a = ldA(aRow, XDP, k * 32, lane);
#pragma unroll
    for (int nt = 0; nt < 4; ++nt)
      acc[nt] = wmma16(a, ldB(WihxH + (size_t)(nBase + nt * 16) * XDP, XDP, k * 32, lane),
                       acc[nt]);
  }
  int n = lane & 15, mhi = (lane & 16) ? 8 : 0;
#pragma unroll
  for (int nt = 0; nt < 4; ++nt) {
    int col = nBase + nt * 16 + n;
    float bb = biasx[col];
#pragma unroll
    for (int r = 0; r < 8; ++r)
      GxH[(rowBase + w * 16 + mhi + r) * GD + col] = (_Float16)(acc[nt][r] + bb);
  }
}

// ---------------------------------------------------------------------------
// K3/K5: persistent single-workgroup LSTM scan. 512 threads = 16 waves
// (4 waves/SIMD -> double VGPR budget, no spills).
// wave w: mTile = w&7 (16 batch rows), jGroup = w>>3 (8 j-tiles of 16 units).
// Each wave computes i/f/g/o tiles for its (mTile, jTile) -> gate quad is
// lane-local. h ping-ponged between two LDS buffers (one barrier/step);
// c state in LDS (f32). Per jt-task: K-outer over 4 gate accumulators so each
// A-frag is loaded once per k and reused x4.
__global__ __launch_bounds__(512, 1) void lstm_seq(
    const _Float16* __restrict__ Gpre,  // (S*B,1024) precomputed x-proj+bias, or null
    const _Float16* __restrict__ Zin,   // (S*B,32) input sequence (decoder), or null
    const _Float16* __restrict__ WihH,  // (1024,32) used when Zin != null
    const _Float16* __restrict__ WhhH,  // (1024,256)
    const float* __restrict__ biasC,    // (1024) or null (folded into Gpre)
    _Float16* __restrict__ outH,        // (S*B,256)
    int reverse) {
  extern __shared__ char smem[];
  _Float16* hbuf0 = (_Float16*)smem;                 // 128*256 f16
  _Float16* hbuf1 = hbuf0 + BATCH * HD;              // 128*256 f16
  float*    cLDS  = (float*)(hbuf1 + BATCH * HD);    // 128*256 f32
  int tid = threadIdx.x, lane = tid & 31, w = tid >> 5;
  int mTile = w & 7, jGroup = w >> 3;  // jGroup in 0..1, 8 j-tiles each
  for (int i = tid; i < BATCH * HD; i += 512) { hbuf0[i] = (_Float16)0.f; cLDS[i] = 0.f; }
  int nn = lane & 15, mhi = (lane & 16) ? 8 : 0;
  int p = 0;

  for (int step = 0; step < SEQL; ++step) {
    int s = reverse ? (SEQL - 1 - step) : step;
    __syncthreads();  // previous step's writes to h[p] visible
    const _Float16* hCur = (p ? hbuf1 : hbuf0) + (size_t)mTile * 16 * HD;
    _Float16* hNxt = p ? hbuf0 : hbuf1;
    v16h az;
    if (Zin) az = ldA(Zin + ((size_t)s * BATCH + mTile * 16) * ZD, ZD, 0, lane);
    for (int jt = 0; jt < 8; ++jt) {
      int jTile = jGroup * 8 + jt;
      v8f acc[4];
#pragma unroll
      for (int g = 0; g < 4; ++g) {
        acc[g] = zero8();
        if (Zin)
          acc[g] = wmma16(az, ldB(WihH + (size_t)(g * HD + jTile * 16) * ZD, ZD, 0, lane),
                          acc[g]);
      }
      for (int k = 0; k < 8; ++k) {
        v16h a = ldA(hCur, HD, k * 32, lane);
#pragma unroll
        for (int g = 0; g < 4; ++g)
          acc[g] = wmma16(a, ldB(WhhH + (size_t)(g * HD + jTile * 16) * HD, HD, k * 32, lane),
                          acc[g]);
      }
      int j = jTile * 16 + nn;
#pragma unroll
      for (int r = 0; r < 8; ++r) {
        int b = mTile * 16 + mhi + r;
        float gi = acc[0][r], gf = acc[1][r], gg = acc[2][r], go = acc[3][r];
        if (Gpre) {
          const _Float16* gp = Gpre + ((size_t)s * BATCH + b) * GD + j;
          gi += (float)gp[0]; gf += (float)gp[HD]; gg += (float)gp[2 * HD]; go += (float)gp[3 * HD];
        }
        if (biasC) {
          gi += biasC[j]; gf += biasC[HD + j]; gg += biasC[2 * HD + j]; go += biasC[3 * HD + j];
        }
        float iv = sigm(gi), fv = sigm(gf), ov = sigm(go), gv = tanhf(gg);
        size_t cidx = (size_t)b * HD + j;
        float cv = fv * cLDS[cidx] + iv * gv;
        cLDS[cidx] = cv;
        float hv = ov * tanhf(cv);
        hNxt[cidx] = (_Float16)hv;
        outH[((size_t)s * BATCH + b) * HD + j] = (_Float16)hv;
      }
    }
    p ^= 1;
  }
}

// ---------------------------------------------------------------------------
// K4: z recurrence. 512 threads = 16 waves, persistent. h_z ping-ponged;
// henc aliases the inactive h_z buffer (lifetimes disjoint); c_z in LDS f32.
__global__ __launch_bounds__(512, 1) void zrec(
    const _Float16* __restrict__ hxH,  // (S*B,256)
    const float* __restrict__ eps,     // (S,B,32)
    const _Float16* __restrict__ W0H, const float* __restrict__ b0,
    const _Float16* __restrict__ WmH, const float* __restrict__ bm,
    const _Float16* __restrict__ WvH, const float* __restrict__ bv,
    const _Float16* __restrict__ WihzH, const _Float16* __restrict__ WhhzH,
    const float* __restrict__ biasz,
    float* __restrict__ outMean, float* __restrict__ outLogvar,
    float* __restrict__ outZ, _Float16* __restrict__ zHg) {
  extern __shared__ char smem[];
  _Float16* hz0 = (_Float16*)smem;                 // 128*256 f16
  _Float16* hz1 = hz0 + BATCH * HD;                // 128*256 f16
  _Float16* zL  = hz1 + BATCH * HD;                // 128*32 f16
  float*    mlv = (float*)(zL + BATCH * ZD);       // 2 * 128*32 f32
  float*    cz  = mlv + 2 * BATCH * ZD;            // 128*256 f32

  int tid = threadIdx.x, lane = tid & 31, w = tid >> 5;
  int mTile = w & 7, jGroup = w >> 3;  // jGroup in 0..1
  int nn = lane & 15, mhi = (lane & 16) ? 8 : 0;
  for (int i = tid; i < BATCH * HD; i += 512) { hz0[i] = (_Float16)0.f; cz[i] = 0.f; }
  int p = 0;

  for (int n = 0; n < SEQL; ++n) {
    __syncthreads();  // previous step's h_z[p] writes visible
    const _Float16* hzCur = (p ? hz1 : hz0) + (size_t)mTile * 16 * HD;
    _Float16* hzNxt = p ? hz0 : hz1;
    _Float16* heH = hzNxt;  // henc lives in the inactive h_z buffer
    // --- (a) henc = tanh([h_x | h_z] @ W0^T + b0): 128x256, K=512 ---
    // K-outer over 4 j-tiles at a time (two halves): A-frag reused x4.
    const _Float16* hxRow = hxH + ((size_t)n * BATCH + mTile * 16) * HD;
    for (int half = 0; half < 2; ++half) {
      v8f acc[4];
#pragma unroll
      for (int q = 0; q < 4; ++q) acc[q] = zero8();
      for (int k = 0; k < 16; ++k) {
        v16h a = (k < 8) ? ldA(hxRow, HD, k * 32, lane)
                         : ldA(hzCur, HD, (k - 8) * 32, lane);
#pragma unroll
        for (int q = 0; q < 4; ++q) {
          int jTile = jGroup * 8 + half * 4 + q;
          acc[q] = wmma16(a, ldB(W0H + (size_t)jTile * 16 * (2 * HD), 2 * HD, k * 32, lane),
                          acc[q]);
        }
      }
#pragma unroll
      for (int q = 0; q < 4; ++q) {
        int j = (jGroup * 8 + half * 4 + q) * 16 + nn;
        float b0j = b0[j];
#pragma unroll
        for (int r = 0; r < 8; ++r) {
          int b = mTile * 16 + mhi + r;
          heH[(size_t)b * HD + j] = (_Float16)tanhf(acc[q][r] + b0j);
        }
      }
    }
    __syncthreads();
    // --- (b) heads: mean / logvar (each 128x32, K=256); 16 tiles per head,
    //     16 waves -> each wave does one tile of each head ---
    for (int sel = 0; sel < 2; ++sel) {
      int mT = w & 7, nT = w >> 3;
      const _Float16* heRow = heH + (size_t)mT * 16 * HD;
      const _Float16* Wh = (sel ? WvH : WmH) + (size_t)nT * 16 * HD;
      const float* bb = sel ? bv : bm;
      float* outP = sel ? outLogvar : outMean;
      v8f acc = zero8();
#pragma unroll
      for (int k = 0; k < 8; ++k)
        acc = wmma16(ldA(heRow, HD, k * 32, lane), ldB(Wh, HD, k * 32, lane), acc);
      int zj = nT * 16 + nn;
      float bz = bb[zj];
#pragma unroll
      for (int r = 0; r < 8; ++r) {
        int b = mT * 16 + mhi + r;
        float v = acc[r] + bz;
        mlv[sel * (BATCH * ZD) + b * ZD + zj] = v;
        outP[(size_t)n * BATCH * ZD + b * ZD + zj] = v;
      }
    }
    __syncthreads();
    // --- (c) z = mean + eps * exp(0.5*logvar) ---
    for (int i = tid; i < BATCH * ZD; i += 512) {
      float zv = mlv[i] + eps[(size_t)n * BATCH * ZD + i] * __expf(0.5f * mlv[BATCH * ZD + i]);
      outZ[(size_t)n * BATCH * ZD + i] = zv;
      _Float16 zh = (_Float16)zv;
      zL[i] = zh;
      zHg[(size_t)n * BATCH * ZD + i] = zh;
    }
    __syncthreads();
    // --- (d) z-LSTM cell: gates = z@Wih_z^T + h_z@Whh_z^T + biasz ---
    v16h az = ldA(zL + (size_t)mTile * 16 * ZD, ZD, 0, lane);
    for (int jt = 0; jt < 8; ++jt) {
      int jTile = jGroup * 8 + jt;
      v8f acc[4];
#pragma unroll
      for (int g = 0; g < 4; ++g) {
        acc[g] = zero8();
        acc[g] = wmma16(az, ldB(WihzH + (size_t)(g * HD + jTile * 16) * ZD, ZD, 0, lane),
                        acc[g]);
      }
      for (int k = 0; k < 8; ++k) {
        v16h a = ldA(hzCur, HD, k * 32, lane);
#pragma unroll
        for (int g = 0; g < 4; ++g)
          acc[g] = wmma16(a, ldB(WhhzH + (size_t)(g * HD + jTile * 16) * HD, HD, k * 32, lane),
                          acc[g]);
      }
      int j = jTile * 16 + nn;
#pragma unroll
      for (int r = 0; r < 8; ++r) {
        int b = mTile * 16 + mhi + r;
        float gi = acc[0][r] + biasz[j];
        float gf = acc[1][r] + biasz[HD + j];
        float gg = acc[2][r] + biasz[2 * HD + j];
        float go = acc[3][r] + biasz[3 * HD + j];
        float iv = sigm(gi), fv = sigm(gf), ov = sigm(go), gv = tanhf(gg);
        size_t cidx = (size_t)b * HD + j;
        float cv = fv * cz[cidx] + iv * gv;
        cz[cidx] = cv;
        hzNxt[cidx] = (_Float16)(ov * tanhf(cv));
      }
    }
    p ^= 1;
  }
}

// ---------------------------------------------------------------------------
// K6: x_rec = exp(h_dec @ Wd^T + bd). One 16x16 tile per wave. N padded to 528.
__global__ __launch_bounds__(256, 1) void head_out(
    const _Float16* __restrict__ hdecH, const _Float16* __restrict__ WdH,
    const float* __restrict__ bd, float* __restrict__ xrec) {
  int tid = threadIdx.x, lane = tid & 31, w = tid >> 5;
  long id = (long)blockIdx.x * 8 + w;   // 4096 mTiles * 33 nTiles
  int mT = (int)(id & 4095);
  int nT = (int)(id >> 12);
  const _Float16* aRow = hdecH + (size_t)mT * 16 * HD;
  v8f acc = zero8();
#pragma unroll
  for (int k = 0; k < 8; ++k)
    acc = wmma16(ldA(aRow, HD, k * 32, lane),
                 ldB(WdH + (size_t)nT * 16 * HD, HD, k * 32, lane), acc);
  int col = nT * 16 + (lane & 15);
  if (col < XD) {
    float bb = bd[col];
    int mhi = (lane & 16) ? 8 : 0;
#pragma unroll
    for (int r = 0; r < 8; ++r)
      xrec[((size_t)mT * 16 + mhi + r) * XD + col] = __expf(acc[r] + bb);
  }
}

// ---------------------------------------------------------------------------
extern "C" void kernel_launch(void* const* d_in, const int* in_sizes, int n_in,
                              void* d_out, int out_size, void* d_ws, size_t ws_size,
                              hipStream_t stream) {
  (void)in_sizes; (void)n_in; (void)out_size; (void)ws_size;
  const float* x     = (const float*)d_in[0];
  const float* eps   = (const float*)d_in[1];
  const float* Wih_x = (const float*)d_in[2];
  const float* Whh_x = (const float*)d_in[3];
  const float* bih_x = (const float*)d_in[4];
  const float* bhh_x = (const float*)d_in[5];
  const float* Wih_z = (const float*)d_in[6];
  const float* Whh_z = (const float*)d_in[7];
  const float* bih_z = (const float*)d_in[8];
  const float* bhh_z = (const float*)d_in[9];
  const float* W0    = (const float*)d_in[10];
  const float* b0    = (const float*)d_in[11];
  const float* Wm    = (const float*)d_in[12];
  const float* bm    = (const float*)d_in[13];
  const float* Wv    = (const float*)d_in[14];
  const float* bv    = (const float*)d_in[15];
  const float* Wih_d = (const float*)d_in[16];
  const float* Whh_d = (const float*)d_in[17];
  const float* bih_d = (const float*)d_in[18];
  const float* bhh_d = (const float*)d_in[19];
  const float* Wd    = (const float*)d_in[20];
  const float* bd    = (const float*)d_in[21];

  // workspace partition (byte offsets, all 256B-aligned)
  char* ws = (char*)d_ws;
  size_t off = 0;
  auto take = [&](size_t bytes) { char* p = ws + off; off += (bytes + 255) & ~(size_t)255; return p; };
  _Float16* WihxH = (_Float16*)take((size_t)GD * XDP * 2);
  _Float16* WhhxH = (_Float16*)take((size_t)GD * HD * 2);
  _Float16* W0H   = (_Float16*)take((size_t)HD * 2 * HD * 2);
  _Float16* WmH   = (_Float16*)take((size_t)ZD * HD * 2);
  _Float16* WvH   = (_Float16*)take((size_t)ZD * HD * 2);
  _Float16* WihzH = (_Float16*)take((size_t)GD * ZD * 2);
  _Float16* WhhzH = (_Float16*)take((size_t)GD * HD * 2);
  _Float16* WihdH = (_Float16*)take((size_t)GD * ZD * 2);
  _Float16* WhhdH = (_Float16*)take((size_t)GD * HD * 2);
  _Float16* WdH   = (_Float16*)take((size_t)WDP * HD * 2);
  float* biasx = (float*)take(GD * 4);
  float* biasz = (float*)take(GD * 4);
  float* biasd = (float*)take(GD * 4);
  _Float16* GxH   = (_Float16*)take((size_t)SEQL * BATCH * GD * 2);
  _Float16* hxH   = (_Float16*)take((size_t)SEQL * BATCH * HD * 2);
  _Float16* zHg   = (_Float16*)take((size_t)SEQL * BATCH * ZD * 2);
  _Float16* hdecH = (_Float16*)take((size_t)SEQL * BATCH * HD * 2);

  float* out = (float*)d_out;
  float* xrec    = out;
  float* outMean = out + (size_t)SEQL * BATCH * XD;
  float* outLv   = outMean + (size_t)SEQL * BATCH * ZD;
  float* outZ    = outLv + (size_t)SEQL * BATCH * ZD;

  auto cvt = [&](const float* s, _Float16* d, int sr, int sc, int dr, int dc) {
    long tot = (long)dr * dc;
    int blocks = (int)((tot + 255) / 256);
    cvt_pad<<<blocks, 256, 0, stream>>>(s, d, sr, sc, dr, dc);
  };
  // K1: weight conversions (zero-padded so GEMM K/N loops are unguarded)
  cvt(Wih_x, WihxH, GD, XD, GD, XDP);
  cvt(Whh_x, WhhxH, GD, HD, GD, HD);
  cvt(W0,    W0H,   HD, 2 * HD, HD, 2 * HD);
  cvt(Wm,    WmH,   ZD, HD, ZD, HD);
  cvt(Wv,    WvH,   ZD, HD, ZD, HD);
  cvt(Wih_z, WihzH, GD, ZD, GD, ZD);
  cvt(Whh_z, WhhzH, GD, HD, GD, HD);
  cvt(Wih_d, WihdH, GD, ZD, GD, ZD);
  cvt(Whh_d, WhhdH, GD, HD, GD, HD);
  cvt(Wd,    WdH,   XD, HD, WDP, HD);
  bias_sum<<<4, 256, 0, stream>>>(bih_x, bhh_x, biasx, GD);
  bias_sum<<<4, 256, 0, stream>>>(bih_z, bhh_z, biasz, GD);
  bias_sum<<<4, 256, 0, stream>>>(bih_d, bhh_d, biasd, GD);

  // K2: precompute encoder input projections (parallel WMMA GEMM)
  gemm_gx<<<dim3(SEQL * BATCH / 128, GD / 64), 256, 128 * XDP * 2, stream>>>(
      x, WihxH, biasx, GxH);

  // K3: backward encoder LSTM (persistent single workgroup, ping-pong h)
  size_t lstm_lds = (size_t)2 * BATCH * HD * 2 + (size_t)BATCH * HD * 4;  // 262144 B
  lstm_seq<<<1, 512, lstm_lds, stream>>>(GxH, nullptr, nullptr, WhhxH,
                                         nullptr, hxH, /*reverse=*/1);

  // K4: z recurrence (ping-pong h_z, henc aliased, c_z in LDS)
  size_t zrec_lds = (size_t)2 * BATCH * HD * 2 + (size_t)BATCH * ZD * 2 +
                    (size_t)2 * BATCH * ZD * 4 + (size_t)BATCH * HD * 4;  // 303104 B
  zrec<<<1, 512, zrec_lds, stream>>>(hxH, eps, W0H, b0, WmH, bm, WvH, bv,
                                     WihzH, WhhzH, biasz,
                                     outMean, outLv, outZ, zHg);

  // K5: forward decoder LSTM
  lstm_seq<<<1, 512, lstm_lds, stream>>>(nullptr, zHg, WihdH, WhhdH,
                                         biasd, hdecH, /*reverse=*/0);

  // K6: output head (parallel WMMA GEMM, exp epilogue)
  long tiles = (long)(SEQL * BATCH / 16) * (WDP / 16);  // 4096*33
  head_out<<<(int)(tiles / 8), 256, 0, stream>>>(hdecH, WdH, bd, xrec);
}
